// ChannelStrip_79688823210250
// MI455X (gfx1250) — matrix operand order repaired and checked by reading, not compiled, not run on hardware
//
#include <hip/hip_runtime.h>
#include <hip/hip_bf16.h>

typedef __attribute__((ext_vector_type(16))) _Float16 v16h;
typedef __attribute__((ext_vector_type(8)))  _Float16 v8h;
typedef __attribute__((ext_vector_type(8)))  float    v8f;

static constexpr int   NFFT = 2048;
static constexpr int   HOP  = 512;
static constexpr int   TLEN = 2880000;             // samples per channel
static constexpr int   CH   = 2;                   // channels
static constexpr int   PAD  = NFFT / 2;            // 1024
static constexpr int   TP   = TLEN + 2 * PAD;      // 2882048 padded length
static constexpr int   NF   = 1 + (TP - NFFT) / HOP;   // 5626 frames
static constexpr float SRF  = 48000.0f;
static constexpr int   SCAN_CHUNK  = 16384;
static constexpr int   SCAN_WARMUP = 65536;
static constexpr int   NCHUNK = (TLEN + SCAN_CHUNK - 1) / SCAN_CHUNK;  // 176

// GEMM tiling: 256-thread block (8 waves), 128 out-dims x 64 frames.
static constexpr int   MT = 128;
static constexpr int   NT = 64;
static constexpr int   SIG_CHUNK = (NT - 1) * HOP + NFFT;          // 34304 halfs
static constexpr int   LPADH = 8;                                  // pad halfs per 512
static constexpr int   ROWH  = HOP + LPADH;                        // 520 halfs padded row
static constexpr int   SMEMH = SIG_CHUNK + LPADH * (SIG_CHUNK / HOP + 1);

__device__ __forceinline__ int lds_off(int t) { return t + LPADH * (t >> 9); }

__device__ __forceinline__ int reflect_idx(int i) {
    if (i < 0) i = -i;
    if (i >= TLEN) i = 2 * TLEN - 2 - i;
    return i;
}

// ---------------- window + wsum ----------------
__global__ void k_window(float* __restrict__ win) {
    int n = blockIdx.x * blockDim.x + threadIdx.x;
    if (n < NFFT)
        win[n] = 0.5f - 0.5f * __cosf(6.2831853071795864f * (float)n / (float)NFFT);
}

__global__ void k_wsum(const float* __restrict__ win, float* __restrict__ wsum) {
    int t = blockIdx.x * blockDim.x + threadIdx.x;
    if (t >= TP) return;
    int jhi = t >> 9;           // t / HOP
    if (jhi > NF - 1) jhi = NF - 1;
    int jlo = 0;
    if (t - (NFFT - 1) > 0) jlo = (t - (NFFT - 1) + (HOP - 1)) >> 9;
    float s = 0.0f;
    for (int j = jlo; j <= jhi; ++j) {
        int n = t - (j << 9);
        if (n >= 0 && n < NFFT) { float w = win[n]; s += w * w; }
    }
    wsum[t] = s;
}

// ---------------- initial reflect pad (f32 audio -> f16 padded) ----------------
__global__ void k_pad_input(const float* __restrict__ audio, _Float16* __restrict__ xh) {
    int i = blockIdx.x * blockDim.x + threadIdx.x;
    if (i >= CH * TP) return;
    int c = i / TP, t = i - c * TP;
    int s = reflect_idx(t - PAD);
    xh[i] = (_Float16)audio[(size_t)c * TLEN + s];
}

// ---------------- per-band circular kernel h = irfft(mask) ----------------
__global__ void k_hker(float* __restrict__ h, const float* __restrict__ gain_db,
                       float f0) {
    int n = blockIdx.x * blockDim.x + threadIdx.x;
    if (n >= NFFT) return;
    float gl = __powf(10.0f, (*gain_db) * 0.05f);
    float s = 0.0f;
    for (int k = 0; k <= NFFT / 2; ++k) {
        float fk = (float)k * (24000.0f / 1024.0f);
        float d = (fk - f0) / f0;                 // q == 1
        float m = 1.0f + (gl - 1.0f) * __expf(-d * d);
        float wgt = (k == 0 || k == NFFT / 2) ? 1.0f : 2.0f;
        int p = (k * n) & (NFFT - 1);             // exact angle mod 2pi
        s += wgt * m * __cosf((6.2831853071795864f / (float)NFFT) * (float)p);
    }
    h[n] = s * (1.0f / (float)NFFT);
}

// ---------------- B[a,b] = w[a] * h[(a-b) mod N] * w[b]  (f16) ----------------
__global__ void k_bmat(const float* __restrict__ win, const float* __restrict__ h,
                       _Float16* __restrict__ Bm) {
    int i = blockIdx.x * blockDim.x + threadIdx.x;
    if (i >= NFFT * NFFT) return;
    int a = i >> 11, b = i & (NFFT - 1);
    Bm[i] = (_Float16)(win[a] * h[(a - b) & (NFFT - 1)] * win[b]);
}

// ---------------- WMMA GEMM + overlap-add ----------------
// 8 waves/block share one LDS-staged signal chunk (the implicit frames matrix);
// wave w computes out-dims [mt*128 + w*16, +16) x frames [nt*64, +64).
// k-loop fully unrolled: every LDS/global access uses a loop-invariant base
// register + compile-time immediate offset; loads grouped ahead of the WMMAs.
__global__ __launch_bounds__(256) void k_gemm_ola(const _Float16* __restrict__ xh,
                                                  const _Float16* __restrict__ Bm,
                                                  float* __restrict__ acc) {
    extern __shared__ _Float16 smem[];
    int mt = blockIdx.x, nt = blockIdx.y, ch = blockIdx.z;
    int tid  = threadIdx.x;
    int wave = tid >> 5, lane = tid & 31;
    const _Float16* xc = xh + (size_t)ch * TP;
    float* accc = acc + (size_t)ch * TP;
    int n0 = nt * NT;
    size_t gbase = (size_t)n0 * HOP;

    // Stage the unique signal samples behind this 64-frame tile (69 KB, padded
    // +16B per 512 halfs so frame-strided reads are bank-conflict-free).
    for (int t = tid * 8; t < SIG_CHUNK; t += 256 * 8) {
        size_t g = gbase + (size_t)t;
        v8h v;
        if (g + 8 <= (size_t)TP) {
            v = *(const v8h*)(xc + g);
        } else {
#pragma unroll
            for (int i = 0; i < 8; ++i) v[i] = (_Float16)0.0f;
        }
        *(v8h*)(smem + lds_off(t)) = v;
    }
    __syncthreads();

    int m0   = mt * MT + wave * 16;
    int half = lane >> 4;                 // 0: lanes 0-15, 1: lanes 16-31
    int mrow = m0 + (lane & 15);          // A-matrix row for this lane
    int col  = lane & 15;                 // B/D column within 16

    // Loop-invariant bases; all per-k addressing folds into immediates.
    const _Float16* apan = Bm + (size_t)mrow * NFFT + half * 8;
    int bb[4];
#pragma unroll
    for (int sub = 0; sub < 4; ++sub)
        bb[sub] = (sub * 16 + col) * ROWH + half * 8;

    v8f cacc[4] = {};                     // 4 x (16x16) accumulators

#pragma unroll
    for (int k0 = 0; k0 < NFFT; k0 += 32) {
        // A fragment (filter matrix) from L2/WGP$: two b128 at immediates.
        v8h alo = *(const v8h*)(apan + k0);
        v8h ahi = *(const v8h*)(apan + k0 + 16);
        v16h a;
#pragma unroll
        for (int i = 0; i < 8; ++i) { a[i] = alo[i]; a[8 + i] = ahi[i]; }

        // All 4 B fragments from LDS first (distinct buffers -> scheduler can
        // overlap next-iteration loads with this iteration's WMMAs).
        const int koff = (k0 & 511) + ROWH * (k0 >> 9);   // compile-time const
        v16h b[4];
#pragma unroll
        for (int sub = 0; sub < 4; ++sub) {
            const _Float16* bp = smem + bb[sub] + koff;
            v8h blo = *(const v8h*)(bp);
            v8h bhi = *(const v8h*)(bp + 16);
#pragma unroll
            for (int i = 0; i < 8; ++i) { b[sub][i] = blo[i]; b[sub][8 + i] = bhi[i]; }
        }

        // Back-to-back WMMAs (disjoint B/C registers: no D->A/B hazard stalls).
#pragma unroll
        for (int sub = 0; sub < 4; ++sub)
            cacc[sub] = __builtin_amdgcn_wmma_f32_16x16x32_f16(
                false, a, false, b[sub], (short)0, cacc[sub], false, false);
    }

    // Epilogue: overlap-add scatter. D layout: lane<16 -> N=col, VGPR v -> M=v(+8).
#pragma unroll
    for (int sub = 0; sub < 4; ++sub) {
        int f = n0 + sub * 16 + col;
        if (f < NF) {
            float* dst = accc + (size_t)f * HOP;
#pragma unroll
            for (int v = 0; v < 8; ++v) {
                int adim = m0 + v + half * 8;
                __hip_atomic_fetch_add(&dst[adim], cacc[sub][v],
                                       __ATOMIC_RELAXED, __HIP_MEMORY_SCOPE_AGENT);
            }
        }
    }
}

// ---------------- normalize + re-pad (feeds next EQ pass, f16) ----------------
__global__ void k_repad(const float* __restrict__ acc, const float* __restrict__ wsum,
                        _Float16* __restrict__ xh) {
    int i = blockIdx.x * blockDim.x + threadIdx.x;
    if (i >= CH * TP) return;
    int c = i / TP, t = i - c * TP;
    int tp = PAD + reflect_idx(t - PAD);
    float ws = wsum[tp];
    float den = ws > 1e-11f ? ws : 1.0f;
    xh[i] = (_Float16)(acc[(size_t)c * TP + tp] / den);
}

// ---------------- normalize + trim (final EQ output, f32) ----------------
__global__ void k_trim(const float* __restrict__ acc, const float* __restrict__ wsum,
                       float* __restrict__ xf) {
    int i = blockIdx.x * blockDim.x + threadIdx.x;
    if (i >= CH * TLEN) return;
    int c = i / TLEN, s = i - c * TLEN;
    int tp = PAD + s;
    float ws = wsum[tp];
    float den = ws > 1e-11f ? ws : 1.0f;
    xf[i] = acc[(size_t)c * TP + tp] / den;
}

// ---------------- compressor gain-reduction input (parallel) ----------------
__global__ void k_gr(const float* __restrict__ xf, float* __restrict__ gr,
                     const float* __restrict__ thr_, const float* __restrict__ ratio_) {
    int i = blockIdx.x * blockDim.x + threadIdx.x;
    if (i >= CH * TLEN) return;
    float thr = *thr_, ratio = *ratio_;
    float aa  = fabsf(xf[i]) + 1e-8f;
    float adb = 20.0f * __log10f(aa);
    gr[i] = fmaxf(adb - thr, 0.0f) * (1.0f - 1.0f / ratio);
}

// ---------------- chunked serial scan + gain + saturation ----------------
__global__ void k_compress(const float* __restrict__ xf, const float* __restrict__ gr,
                           float* __restrict__ out,
                           const float* __restrict__ att_, const float* __restrict__ rel_,
                           const float* __restrict__ mk_, const float* __restrict__ sat_) {
    int idx = blockIdx.x * blockDim.x + threadIdx.x;
    if (idx >= CH * NCHUNK) return;
    int c  = idx / NCHUNK;
    int ck = idx - c * NCHUNK;

    float a_att = 1.0f - __expf(-1.0f / ((*att_) * SRF));
    float a_rel = 1.0f - __expf(-1.0f / ((*rel_) * SRF));
    float mk = *mk_, sat = *sat_;

    const float* g = gr + (size_t)c * TLEN;
    const float* x = xf + (size_t)c * TLEN;
    float* o = out + (size_t)c * TLEN;

    long start = (long)ck * SCAN_CHUNK;
    long end   = start + SCAN_CHUNK; if (end > TLEN) end = TLEN;
    long ws    = start - SCAN_WARMUP; if (ws < 1) ws = 1;

    if (start == 0) {   // t = 0: gain state is exactly 0
        float xv = x[0];
        float aa = fabsf(xv) + 1e-8f;
        float gl = __powf(10.0f, mk * 0.05f);
        float sgn = xv > 0.0f ? 1.0f : (xv < 0.0f ? -1.0f : 0.0f);
        float y = sgn * aa * gl;
        o[0] = sat > 1.0f ? tanhf(y * sat) / sat : y;
    }

    float prev = 0.0f;  // state contribution decays over the warm-up (contractive IIR)
    for (long t = ws; t < end; ++t) {
        float gv = g[t];
        float alpha = gv > prev ? a_att : a_rel;
        prev += alpha * (gv - prev);
        if (t >= start) {
            float xv = x[t];
            float aa = fabsf(xv) + 1e-8f;
            float gl = __powf(10.0f, (mk - prev) * 0.05f);
            float sgn = xv > 0.0f ? 1.0f : (xv < 0.0f ? -1.0f : 0.0f);
            float y = sgn * aa * gl;
            o[t] = sat > 1.0f ? tanhf(y * sat) / sat : y;
        }
    }
}

// ---------------- launch ----------------
static inline size_t align256(size_t x) { return (x + 255) & ~(size_t)255; }

extern "C" void kernel_launch(void* const* d_in, const int* in_sizes, int n_in,
                              void* d_out, int out_size, void* d_ws, size_t ws_size,
                              hipStream_t stream) {
    const float* audio = (const float*)d_in[0];
    const float* g_eq[4] = { (const float*)d_in[1], (const float*)d_in[2],
                             (const float*)d_in[3], (const float*)d_in[4] };
    const float* thr   = (const float*)d_in[5];
    const float* ratio = (const float*)d_in[6];
    const float* att   = (const float*)d_in[7];
    const float* rel   = (const float*)d_in[8];
    const float* mkup  = (const float*)d_in[9];
    const float* sat   = (const float*)d_in[10];
    float* out = (float*)d_out;

    // workspace carve-up
    char* p = (char*)d_ws;
    size_t off = 0;
    _Float16* xh   = (_Float16*)(p + off); off = align256(off + (size_t)CH * TP * sizeof(_Float16));
    float*    acc  = (float*)   (p + off); off = align256(off + (size_t)CH * TP * sizeof(float));
    float*    wsum = (float*)   (p + off); off = align256(off + (size_t)TP * sizeof(float));
    float*    win  = (float*)   (p + off); off = align256(off + (size_t)NFFT * sizeof(float));
    float*    hker = (float*)   (p + off); off = align256(off + (size_t)NFFT * sizeof(float));
    _Float16* Bm   = (_Float16*)(p + off); off = align256(off + (size_t)NFFT * NFFT * sizeof(_Float16));
    float*    xf   = (float*)   (p + off); off = align256(off + (size_t)CH * TLEN * sizeof(float));
    float*    gr   = (float*)   (p + off); off = align256(off + (size_t)CH * TLEN * sizeof(float));

    const int BS = 256;
    dim3 gemm_grid(NFFT / MT, (NF + NT - 1) / NT, CH);
    size_t gemm_lds = (size_t)SMEMH * sizeof(_Float16);

    k_window<<<(NFFT + BS - 1) / BS, BS, 0, stream>>>(win);
    k_wsum<<<(TP + BS - 1) / BS, BS, 0, stream>>>(win, wsum);
    k_pad_input<<<(CH * TP + BS - 1) / BS, BS, 0, stream>>>(audio, xh);

    const float band_freq[4] = { 100.0f, 500.0f, 3000.0f, 10000.0f };
    for (int b = 0; b < 4; ++b) {
        k_hker<<<(NFFT + BS - 1) / BS, BS, 0, stream>>>(hker, g_eq[b], band_freq[b]);
        k_bmat<<<(NFFT * NFFT + BS - 1) / BS, BS, 0, stream>>>(win, hker, Bm);
        hipMemsetAsync(acc, 0, (size_t)CH * TP * sizeof(float), stream);
        k_gemm_ola<<<gemm_grid, 256, gemm_lds, stream>>>(xh, Bm, acc);
        if (b < 3)
            k_repad<<<(CH * TP + BS - 1) / BS, BS, 0, stream>>>(acc, wsum, xh);
        else
            k_trim<<<(CH * TLEN + BS - 1) / BS, BS, 0, stream>>>(acc, wsum, xf);
    }

    k_gr<<<(CH * TLEN + BS - 1) / BS, BS, 0, stream>>>(xf, gr, thr, ratio);
    k_compress<<<(CH * NCHUNK + 31) / 32, 32, 0, stream>>>(xf, gr, out, att, rel, mkup, sat);
}
